// CosineSimilarity2DVectors_38233798869554
// MI455X (gfx1250) — matrix operand order repaired and checked by reading, not compiled, not run on hardware
//
#include <hip/hip_runtime.h>
#include <math.h>
#include <stdint.h>

// CDNA5 / gfx1250, wave32.
// C[n,m] = sum_d A[n,d]*B[m,d], optionally / (sqrt(|a_n|*|b_m|) + eps).
// Pipeline:
//   1) split A,B (fp32) once into bf16 hi/lo planes in d_ws (memory-bound).
//   2) row norms of A,B into d_ws.
//   3) 128x128-block GEMM, bf16x3 (hi*hi + hi*lo + lo*hi) on v_wmma
//      16x16x32_bf16, tiles double-buffered through LDS via async-to-LDS
//      loads (ASYNCcnt).

typedef __attribute__((ext_vector_type(16))) __bf16 v16bf;
typedef __attribute__((ext_vector_type(8)))  __bf16 v8bf;
typedef __attribute__((ext_vector_type(8)))  float  v8f;
typedef __attribute__((ext_vector_type(4)))  float  f32x4;
typedef __attribute__((ext_vector_type(4)))  int    v4i;

typedef __attribute__((address_space(1))) v4i gv4i;  // global v4i
typedef __attribute__((address_space(3))) v4i lv4i;  // LDS v4i

#define EPS 1e-6f
#define LDS_STRIDE 40   // elements; 80B row stride -> conflict-free b128 reads

#if __has_builtin(__builtin_amdgcn_global_load_async_to_lds_b128)
#define HAVE_ASYNC_LDS 1
#endif

// ---- 16-byte global -> LDS copy (async if the toolchain exposes it) -------
__device__ __forceinline__ void copy16B(const __bf16* gsrc, __bf16* ldst) {
#if defined(HAVE_ASYNC_LDS)
  __builtin_amdgcn_global_load_async_to_lds_b128(
      (gv4i*)(uintptr_t)gsrc,
      (lv4i*)(uint32_t)(uintptr_t)ldst,
      0, 0);
#else
  *(v8bf*)ldst = *(const v8bf*)gsrc;
#endif
}

#if defined(HAVE_ASYNC_LDS)
#if __has_builtin(__builtin_amdgcn_s_wait_asynccnt)
#define WAIT_ASYNC(n) __builtin_amdgcn_s_wait_asynccnt(n)
#else
#define WAIT_ASYNC(n) asm volatile("s_wait_asynccnt %0" ::"i"(n) : "memory")
#endif
#else
#define WAIT_ASYNC(n) ((void)0)
#endif

// ---- assemble a 16-element bf16 fragment from two 16B LDS chunks ----------
__device__ __forceinline__ v16bf frag16(const __bf16* p, int off0, int off1) {
  v8bf x = *(const v8bf*)(p + off0);
  v8bf y = *(const v8bf*)(p + off1);
  return __builtin_shufflevector(x, y, 0, 1, 2, 3, 4, 5, 6, 7,
                                 8, 9, 10, 11, 12, 13, 14, 15);
}

// ---------------------------------------------------------------------------
// fp32 -> bf16 hi/lo split planes.  8 elements per thread.
// ---------------------------------------------------------------------------
__global__ __launch_bounds__(256)
void split_kernel(const float* __restrict__ X, __bf16* __restrict__ hi,
                  __bf16* __restrict__ lo, size_t n) {
  size_t idx = ((size_t)blockIdx.x * 256 + threadIdx.x) * 8;
  if (idx >= n) return;
  f32x4 x0 = *reinterpret_cast<const f32x4*>(X + idx);
  f32x4 x1 = *reinterpret_cast<const f32x4*>(X + idx + 4);
  v8bf h, l;
#pragma unroll
  for (int j = 0; j < 4; ++j) {
    float f = x0[j];
    __bf16 hb = (__bf16)f;
    h[j] = hb;
    l[j] = (__bf16)(f - (float)hb);
  }
#pragma unroll
  for (int j = 0; j < 4; ++j) {
    float f = x1[j];
    __bf16 hb = (__bf16)f;
    h[4 + j] = hb;
    l[4 + j] = (__bf16)(f - (float)hb);
  }
  *(v8bf*)(hi + idx) = h;
  *(v8bf*)(lo + idx) = l;
}

// ---------------------------------------------------------------------------
// Row norms: one wave per row.
// ---------------------------------------------------------------------------
__global__ __launch_bounds__(256)
void row_norm_kernel(const float* __restrict__ X, float* __restrict__ out,
                     int nrows, int D) {
  int row  = blockIdx.x * 8 + (threadIdx.x >> 5);
  int lane = threadIdx.x & 31;
  if (row >= nrows) return;
  const float* p = X + (size_t)row * D;
  float s = 0.f;
  for (int i = lane; i < D; i += 32) {
    float v = p[i];
    s += v * v;
  }
#pragma unroll
  for (int off = 16; off > 0; off >>= 1)
    s += __shfl_down(s, off, 32);
  if (lane == 0) out[row] = sqrtf(s);
}

// ---------------------------------------------------------------------------
// Main GEMM (pre-split planes, LDS staged, double buffered).
// Block = 256 threads = 8 waves (2 x 4). Block tile 128x128; wave tile 64x32.
// Assumes N,M % 128 == 0, D % 32 == 0 (8192/8192/512).
// ---------------------------------------------------------------------------
__global__ __launch_bounds__(256, 1)
void gemm_lds(const __bf16* __restrict__ Ahi, const __bf16* __restrict__ Alo,
              const __bf16* __restrict__ Bhi, const __bf16* __restrict__ Blo,
              const float* __restrict__ n1, const float* __restrict__ n2,
              const int* __restrict__ normalize, float* __restrict__ C,
              int N, int M, int D) {
  __shared__ __attribute__((aligned(16))) __bf16 lds[2][4][128 * LDS_STRIDE];

  const int tid   = threadIdx.x;
  const int lane  = tid & 31;
  const int wave  = tid >> 5;
  const int r16   = lane & 15;
  const int kHalf = lane >> 4;
  const int waveM = wave & 1;   // 2 waves along output rows
  const int waveN = wave >> 1;  // 4 waves along output cols

  const int rowBase = blockIdx.y * 128;
  const int colBase = blockIdx.x * 128;

  const int aoff = 8 * kHalf;   // A frag: K = {aoff..+7, 16+aoff..+7}
  const int boff = 8 * kHalf;   // B frag: K = {boff..+7, 16+boff..+7}

  const __bf16* gplanes[4] = {Ahi, Alo, Bhi, Blo};

  // Staging: per plane, 128 rows x 32 k = 512 x 16B chunks; 2 chunks/thread.
  // chunk c: row = c>>2, q = c&3 (16B = 8 elements at k-offset q*8).
  auto stage = [&](int buf, int kk) {
#pragma unroll
    for (int p = 0; p < 4; ++p) {
      const __bf16* gbase = gplanes[p];
      const int rb = (p < 2) ? rowBase : colBase;
#pragma unroll
      for (int c = 0; c < 2; ++c) {
        const int chunk = tid + c * 256;
        const int row = chunk >> 2;
        const int q   = chunk & 3;
        copy16B(gbase + (size_t)(rb + row) * D + kk + q * 8,
                &lds[buf][p][row * LDS_STRIDE + q * 8]);
      }
    }
  };

  v8f acc[4][2];
#pragma unroll
  for (int i = 0; i < 4; ++i)
#pragma unroll
    for (int j = 0; j < 2; ++j) acc[i][j] = v8f{};

  stage(0, 0);
  int cur = 0;
  for (int kk = 0; kk < D; kk += 32) {
    if (kk + 32 < D) {
      stage(cur ^ 1, kk + 32);
      WAIT_ASYNC(8);            // allow the 8 just-issued to stay outstanding
    } else {
      WAIT_ASYNC(0);
    }
    __syncthreads();

    const __bf16* pAh = &lds[cur][0][0];
    const __bf16* pAl = &lds[cur][1][0];
    const __bf16* pBh = &lds[cur][2][0];
    const __bf16* pBl = &lds[cur][3][0];

    v16bf ahi[4], alo[4], bhi[2], blo[2];
#pragma unroll
    for (int i = 0; i < 4; ++i) {
      const int o = (waveM * 64 + i * 16 + r16) * LDS_STRIDE;
      ahi[i] = frag16(pAh + o, aoff, 16 + aoff);
      alo[i] = frag16(pAl + o, aoff, 16 + aoff);
    }
#pragma unroll
    for (int j = 0; j < 2; ++j) {
      const int o = (waveN * 32 + j * 16 + r16) * LDS_STRIDE;
      bhi[j] = frag16(pBh + o, boff, 16 + boff);
      blo[j] = frag16(pBl + o, boff, 16 + boff);
    }

#pragma unroll
    for (int i = 0; i < 4; ++i)
#pragma unroll
      for (int j = 0; j < 2; ++j) {
        acc[i][j] = __builtin_amdgcn_wmma_f32_16x16x32_bf16(
            false, ahi[i], false, bhi[j], (short)0, acc[i][j], false, false);
        acc[i][j] = __builtin_amdgcn_wmma_f32_16x16x32_bf16(
            false, ahi[i], false, blo[j], (short)0, acc[i][j], false, false);
        acc[i][j] = __builtin_amdgcn_wmma_f32_16x16x32_bf16(
            false, alo[i], false, bhi[j], (short)0, acc[i][j], false, false);
      }

    __syncthreads();  // everyone done reading lds[cur] before it is restaged
    cur ^= 1;
  }

  // Epilogue. C/D layout: VGPR r -> M = r + 8*(lane>>4), N = lane&15.
  const bool doNorm = (*normalize) != 0;
#pragma unroll
  for (int i = 0; i < 4; ++i) {
    const int row0 = rowBase + waveM * 64 + i * 16 + 8 * kHalf;
#pragma unroll
    for (int j = 0; j < 2; ++j) {
      const int col = colBase + waveN * 32 + j * 16 + r16;
      const float nb = doNorm ? n2[col] : 1.f;
#pragma unroll
      for (int r = 0; r < 8; ++r) {
        const int row = row0 + r;
        float v = acc[i][j][r];
        if (doNorm) v = v / (sqrtf(n1[row] * nb) + EPS);
        C[(size_t)row * M + col] = v;
      }
    }
  }
}

// ---------------------------------------------------------------------------
// Fallback GEMM (inline fp32->bf16 split, no ws planes needed).
// ---------------------------------------------------------------------------
__device__ __forceinline__ void split8g(const float* __restrict__ q,
                                        v16bf& hi, v16bf& lo, int base) {
#pragma unroll
  for (int i = 0; i < 8; i += 4) {
    f32x4 x = *reinterpret_cast<const f32x4*>(q + i);
#pragma unroll
    for (int j = 0; j < 4; ++j) {
      float f = x[j];
      __bf16 h = (__bf16)f;
      hi[base + i + j] = h;
      lo[base + i + j] = (__bf16)(f - (float)h);
    }
  }
}

__global__ __launch_bounds__(256, 1)
void gemm_inline(const float* __restrict__ A, const float* __restrict__ B,
                 const float* __restrict__ n1, const float* __restrict__ n2,
                 const int* __restrict__ normalize, float* __restrict__ C,
                 int N, int M, int D) {
  const int lane  = threadIdx.x & 31;
  const int wave  = threadIdx.x >> 5;
  const int r16   = lane & 15;
  const int kHalf = lane >> 4;

  const int rowBase = blockIdx.y * 64;
  const int colBase = blockIdx.x * 256 + wave * 32;

  v8f acc[4][2];
#pragma unroll
  for (int i = 0; i < 4; ++i)
#pragma unroll
    for (int j = 0; j < 2; ++j) acc[i][j] = v8f{};

  const float* aptr[4];
  const float* bptr[2];
#pragma unroll
  for (int i = 0; i < 4; ++i)
    aptr[i] = A + (size_t)(rowBase + i * 16 + r16) * D;
#pragma unroll
  for (int j = 0; j < 2; ++j)
    bptr[j] = B + (size_t)(colBase + j * 16 + r16) * D;

  const int aoff = 8 * kHalf;
  const int boff = 8 * kHalf;

  for (int kk = 0; kk < D; kk += 32) {
    v16bf ahi[4], alo[4], bhi[2], blo[2];
#pragma unroll
    for (int i = 0; i < 4; ++i) {
      split8g(aptr[i] + aoff, ahi[i], alo[i], 0);
      split8g(aptr[i] + 16 + aoff, ahi[i], alo[i], 8);
      aptr[i] += 32;
    }
#pragma unroll
    for (int j = 0; j < 2; ++j) {
      split8g(bptr[j] + boff, bhi[j], blo[j], 0);
      split8g(bptr[j] + 16 + boff, bhi[j], blo[j], 8);
      bptr[j] += 32;
    }
#pragma unroll
    for (int i = 0; i < 4; ++i)
#pragma unroll
      for (int j = 0; j < 2; ++j) {
        acc[i][j] = __builtin_amdgcn_wmma_f32_16x16x32_bf16(
            false, ahi[i], false, bhi[j], (short)0, acc[i][j], false, false);
        acc[i][j] = __builtin_amdgcn_wmma_f32_16x16x32_bf16(
            false, ahi[i], false, blo[j], (short)0, acc[i][j], false, false);
        acc[i][j] = __builtin_amdgcn_wmma_f32_16x16x32_bf16(
            false, alo[i], false, bhi[j], (short)0, acc[i][j], false, false);
      }
  }

  const bool doNorm = (*normalize) != 0;
#pragma unroll
  for (int i = 0; i < 4; ++i) {
    const int row0 = rowBase + i * 16 + 8 * kHalf;
#pragma unroll
    for (int j = 0; j < 2; ++j) {
      const int col = colBase + j * 16 + r16;
      const float nb = doNorm ? n2[col] : 1.f;
#pragma unroll
      for (int r = 0; r < 8; ++r) {
        const int row = row0 + r;
        float v = acc[i][j][r];
        if (doNorm) v = v / (sqrtf(n1[row] * nb) + EPS);
        C[(size_t)row * M + col] = v;
      }
    }
  }
}

// ---------------------------------------------------------------------------
extern "C" void kernel_launch(void* const* d_in, const int* in_sizes, int n_in,
                              void* d_out, int out_size, void* d_ws, size_t ws_size,
                              hipStream_t stream) {
  const float* A = (const float*)d_in[0];
  const float* B = (const float*)d_in[1];
  const int* normalize = (const int*)d_in[2];
  float* C = (float*)d_out;

  const int D = 512;
  const int N = in_sizes[0] / D;
  const int M = in_sizes[1] / D;

  char* ws = (char*)d_ws;
  float* n1 = (float*)ws;  // N floats
  float* n2 = n1 + N;      // M floats

  row_norm_kernel<<<dim3((N + 7) / 8), 256, 0, stream>>>(A, n1, N, D);
  row_norm_kernel<<<dim3((M + 7) / 8), 256, 0, stream>>>(B, n2, M, D);

  const size_t normBytes  = (size_t)(N + M) * sizeof(float);
  const size_t planeBytes = (size_t)(N + M) * D * 2 * sizeof(__bf16);

  if (ws_size >= normBytes + planeBytes) {
    __bf16* Ahi = (__bf16*)(ws + normBytes);
    __bf16* Alo = Ahi + (size_t)N * D;
    __bf16* Bhi = Alo + (size_t)N * D;
    __bf16* Blo = Bhi + (size_t)M * D;

    const size_t nA = (size_t)N * D, nB = (size_t)M * D;
    split_kernel<<<dim3((unsigned)(nA / (256 * 8))), 256, 0, stream>>>(A, Ahi, Alo, nA);
    split_kernel<<<dim3((unsigned)(nB / (256 * 8))), 256, 0, stream>>>(B, Bhi, Blo, nB);

    dim3 grid(M / 128, N / 128);
    gemm_lds<<<grid, 256, 0, stream>>>(Ahi, Alo, Bhi, Blo, n1, n2, normalize,
                                       C, N, M, D);
  } else {
    dim3 grid(M / 256, N / 64);
    gemm_inline<<<grid, 256, 0, stream>>>(A, B, n1, n2, normalize, C, N, M, D);
  }
}
